// HeadTransformerBlock_64441689309942
// MI455X (gfx1250) — matrix operand order repaired and checked by reading, not compiled, not run on hardware
//
#include <hip/hip_runtime.h>

#define S_LEN 2048
#define D_DIM 1024
#define H_NUM 16
#define DH_DIM 64
#define F_DIM 4096

typedef __attribute__((ext_vector_type(16))) __bf16 v16bf;
typedef __attribute__((ext_vector_type(8)))  __bf16 v8bf;
typedef __attribute__((ext_vector_type(8)))  float  v8f;
typedef __attribute__((ext_vector_type(4)))  unsigned int u32x4;
typedef __attribute__((ext_vector_type(8)))  unsigned int u32x8;

__device__ __forceinline__ v8f zero8() {
  v8f z;
#pragma unroll
  for (int i = 0; i < 8; i++) z[i] = 0.f;
  return z;
}

__device__ __forceinline__ float gelu_f(float x) {
  const float c0 = 0.7978845608028654f; // sqrt(2/pi)
  return 0.5f * x * (1.f + tanhf(c0 * (x + 0.044715f * x * x * x)));
}

// ---------------------------------------------------------------------------
// Tensor Data Mover: issue a 2D tile load (global -> LDS) via a D# descriptor
// built per CDNA5 ISA (group0: 4 SGPRs, group1: 8 SGPRs).
// data_size = 4B units. Tile width == tensor width (rows always in-bounds).
// Caller must be wave-uniform; completion via s_wait_tensorcnt (in-order).
// ---------------------------------------------------------------------------
__device__ __forceinline__ void tdm_load_2d(unsigned lds_byte_addr,
                                            const void* gptr,
                                            unsigned row_bytes,    // tile width in bytes (mult of 4)
                                            unsigned rows,         // tile height
                                            unsigned pitch_bytes)  // tensor row pitch in bytes
{
  const unsigned long long ga = (unsigned long long)(size_t)gptr;
  const unsigned units  = row_bytes >> 2;   // width in 4B units
  const unsigned stride = pitch_bytes >> 2; // pitch in 4B units

  // group 0: count=1 | lds_addr | global_addr[56:0] | type=2
  const unsigned w0 = 1u;
  const unsigned w1 = lds_byte_addr;
  const unsigned w2 = (unsigned)ga;
  const unsigned w3 = ((unsigned)(ga >> 32) & 0x01FFFFFFu) | 0x80000000u;
  // group 1: wg_mask=0 (not in cluster), data_size=2(4B), no pad/iterate/barrier
  const unsigned g1w0 = 2u << 16;
  const unsigned g1w1 = (units & 0xFFFFu) << 16;                    // tensor_dim0[15:0]
  const unsigned g1w2 = (units >> 16) | ((rows & 0xFFFFu) << 16);   // tdim0 hi | tensor_dim1 lo
  const unsigned g1w3 = (rows >> 16) | ((units & 0xFFFFu) << 16);   // tdim1 hi | tile_dim0
  const unsigned g1w4 = rows & 0xFFFFu;                             // tile_dim1 (tile_dim2=0)
  const unsigned g1w5 = stride;                                     // tensor_dim0_stride lo
  const unsigned g1w6 = 0u;                                         // stride hi | dim1_stride lo
  const unsigned g1w7 = 0u;

  u32x4 g0;
  g0[0] = (unsigned)__builtin_amdgcn_readfirstlane((int)w0);
  g0[1] = (unsigned)__builtin_amdgcn_readfirstlane((int)w1);
  g0[2] = (unsigned)__builtin_amdgcn_readfirstlane((int)w2);
  g0[3] = (unsigned)__builtin_amdgcn_readfirstlane((int)w3);
  u32x8 g1;
  g1[0] = (unsigned)__builtin_amdgcn_readfirstlane((int)g1w0);
  g1[1] = (unsigned)__builtin_amdgcn_readfirstlane((int)g1w1);
  g1[2] = (unsigned)__builtin_amdgcn_readfirstlane((int)g1w2);
  g1[3] = (unsigned)__builtin_amdgcn_readfirstlane((int)g1w3);
  g1[4] = (unsigned)__builtin_amdgcn_readfirstlane((int)g1w4);
  g1[5] = (unsigned)__builtin_amdgcn_readfirstlane((int)g1w5);
  g1[6] = (unsigned)__builtin_amdgcn_readfirstlane((int)g1w6);
  g1[7] = (unsigned)__builtin_amdgcn_readfirstlane((int)g1w7);

  asm volatile("tensor_load_to_lds %0, %1" :: "s"(g0), "s"(g1) : "memory");
}

// ---------------------------------------------------------------------------
// fp32 -> bf16 convert + transpose into Bt[N][K] (K contiguous per column).
// ---------------------------------------------------------------------------
__global__ __launch_bounds__(256)
void tcvt_kernel(const float* __restrict__ in, __bf16* __restrict__ out,
                 int K, int N, int head_blocked) {
  const size_t total = (size_t)K * (size_t)N;
  for (size_t idx = (size_t)blockIdx.x * 256ull + threadIdx.x; idx < total;
       idx += (size_t)gridDim.x * 256ull) {
    const int nn = (int)(idx / (size_t)K);
    const int kk = (int)(idx - (size_t)nn * (size_t)K);
    size_t src;
    if (head_blocked)
      src = (size_t)(nn >> 6) * ((size_t)K * 64) + (size_t)kk * 64 + (size_t)(nn & 63);
    else
      src = (size_t)kk * (size_t)N + (size_t)nn;
    out[idx] = (__bf16)in[src];
  }
}

// ---------------------------------------------------------------------------
// Rowwise LayerNorm, fp32 in -> bf16 out. One block (256 thr) per row.
// ---------------------------------------------------------------------------
__global__ __launch_bounds__(256)
void ln_kernel(const float* __restrict__ x, const float* __restrict__ w,
               const float* __restrict__ b, __bf16* __restrict__ out, int D) {
  const int row = blockIdx.x;
  const float* xr = x + (size_t)row * D;
  __shared__ float sh[8];
  const int tid = threadIdx.x, lane = tid & 31, wid = tid >> 5;

  float s = 0.f;
  for (int i = tid; i < D; i += 256) s += xr[i];
#pragma unroll
  for (int o = 16; o > 0; o >>= 1) s += __shfl_xor(s, o, 32);
  if (lane == 0) sh[wid] = s;
  __syncthreads();
  float tot = 0.f;
#pragma unroll
  for (int i = 0; i < 8; i++) tot += sh[i];
  const float mu = tot / (float)D;
  __syncthreads();

  float v2 = 0.f;
  for (int i = tid; i < D; i += 256) { float d = xr[i] - mu; v2 += d * d; }
#pragma unroll
  for (int o = 16; o > 0; o >>= 1) v2 += __shfl_xor(v2, o, 32);
  if (lane == 0) sh[wid] = v2;
  __syncthreads();
  float tv = 0.f;
#pragma unroll
  for (int i = 0; i < 8; i++) tv += sh[i];
  const float rstd = rsqrtf(tv / (float)D + 1e-5f);

  for (int i = tid; i < D; i += 256)
    out[(size_t)row * D + i] = (__bf16)((xr[i] - mu) * rstd * w[i] + b[i]);
}

// ---------------------------------------------------------------------------
// GEMM: C[M,N] = A[M,K](bf16 row-major) x Bt[N,K](bf16, K contiguous per col)
// Block tile 128x64 (8 waves, 4x2); each wave computes 32x32 (2x2 WMMA tiles).
// K-loop double-buffers A/B tiles in LDS via the Tensor Data Mover: tile k+1's
// DMA is issued before computing on tile k, then s_wait_tensorcnt 2 retires
// only the two oldest (in-order) transfers. WMMA fed by ds_load_b128.
// Fragment layouts per CDNA5 ISA 7.12.2:
//   A: lane (m,g) holds row m;  element e -> K = e + 8g + 8*(e>>3)
//   B: lane (n,g) holds col n;  element e -> K = e + 8g + 8*(e>>3)
//   C: element r -> row r + 8g, col n
// ---------------------------------------------------------------------------
template <int ACT, int OUT_BF16, int OUT_F32, int ADD_RESID>
__global__ __launch_bounds__(256)
void gemm_bf16_wmma(const __bf16* __restrict__ A, const __bf16* __restrict__ Bt,
                    const float* __restrict__ bias, const float* __restrict__ resid,
                    __bf16* __restrict__ outb, float* __restrict__ outf,
                    int M, int N, int K) {
  constexpr int BM = 128, BN = 64, BK = 32;
  __shared__ __bf16 Abuf[2][BM * BK];  // row-major, 64B per row
  __shared__ __bf16 Bbuf[2][BN * BK];  // col-of-C major, 64B per row

  const int nblk = N / BN;
  const int mb = blockIdx.x / nblk;
  const int nb = blockIdx.x - mb * nblk;
  const int row0 = mb * BM, col0 = nb * BN;

  const int tid = (int)threadIdx.x;
  const int wave = tid >> 5;
  const int lane = tid & 31;
  const int wm = wave >> 1, wn = wave & 1;  // 4x2 wave grid
  const int n = lane & 15, g = lane >> 4;

  v8f acc[2][2];
#pragma unroll
  for (int t = 0; t < 2; t++)
#pragma unroll
    for (int u = 0; u < 2; u++) acc[t][u] = zero8();

  const __bf16* Ablk = A + (size_t)row0 * K;
  const __bf16* Bblk = Bt + (size_t)col0 * K;
  const int nk = K / BK;

  // Prologue: kick off DMA for tile 0 (TDM ignores EXEC: single wave issues).
  if (wave == 0) {
    tdm_load_2d((unsigned)(size_t)(&Abuf[0][0]), Ablk, BK * 2, BM, (unsigned)K * 2);
    tdm_load_2d((unsigned)(size_t)(&Bbuf[0][0]), Bblk, BK * 2, BN, (unsigned)K * 2);
  }

  for (int kt = 0; kt < nk; kt++) {
    const int pb = kt & 1;
    if (wave == 0) {
      if (kt + 1 < nk) {
        // Issue next tile's DMA into the alternate buffer (last read two
        // barriers ago), then wait for the two OLDEST transfers (current tile).
        const int k1 = (kt + 1) * BK;
        tdm_load_2d((unsigned)(size_t)(&Abuf[1 - pb][0]), Ablk + k1, BK * 2, BM, (unsigned)K * 2);
        tdm_load_2d((unsigned)(size_t)(&Bbuf[1 - pb][0]), Bblk + k1, BK * 2, BN, (unsigned)K * 2);
        __builtin_amdgcn_s_wait_tensorcnt(2);
      } else {
        __builtin_amdgcn_s_wait_tensorcnt(0);
      }
    }
    __syncthreads();  // publish current buffer to all waves

    v16bf af[2], bfr[2];
#pragma unroll
    for (int t = 0; t < 2; t++) {
      const __bf16* ar = &Abuf[pb][0] + (size_t)(wm * 32 + t * 16 + n) * BK;
      v8bf a0 = *(const v8bf*)(ar + 8 * g);
      v8bf a1 = *(const v8bf*)(ar + 16 + 8 * g);
      const __bf16* br = &Bbuf[pb][0] + (size_t)(wn * 32 + t * 16 + n) * BK;
      v8bf b0 = *(const v8bf*)(br + 8 * g);
      v8bf b1 = *(const v8bf*)(br + 16 + 8 * g);
#pragma unroll
      for (int i = 0; i < 8; i++) {
        af[t][i] = a0[i];  af[t][i + 8] = a1[i];
        bfr[t][i] = b0[i]; bfr[t][i + 8] = b1[i];
      }
    }
#pragma unroll
    for (int t = 0; t < 2; t++)
#pragma unroll
      for (int u = 0; u < 2; u++)
        acc[t][u] = __builtin_amdgcn_wmma_f32_16x16x32_bf16(
            false, af[t], false, bfr[u], (short)0, acc[t][u], false, false);
    __syncthreads();  // all reads of this buffer done before it is re-filled
  }

#pragma unroll
  for (int u = 0; u < 2; u++) {
    const int col = col0 + wn * 32 + u * 16 + n;
    const float bc = bias ? bias[col] : 0.f;
#pragma unroll
    for (int t = 0; t < 2; t++) {
#pragma unroll
      for (int r = 0; r < 8; r++) {
        const int row = row0 + wm * 32 + t * 16 + r + 8 * g;
        float v = acc[t][u][r] + bc;
        if (ACT == 1) v = gelu_f(v);
        if (ADD_RESID) v += resid[(size_t)row * N + col];
        if (OUT_F32) outf[(size_t)row * N + col] = v;
        if (OUT_BF16) outb[(size_t)row * N + col] = (__bf16)v;
      }
    }
  }
}

// ---------------------------------------------------------------------------
// Causal attention, one wave per (16-query block, head). Online softmax over
// 32-key chunks; scores and z accumulation via WMMA; P and transposed-V tiles
// staged in LDS. Applies per-head hard gate from mask_logits.
// q,k,v,z are bf16 [S, H*DH] row-major.
// ---------------------------------------------------------------------------
__global__ __launch_bounds__(32)
void attn_kernel(const __bf16* __restrict__ q, const __bf16* __restrict__ k,
                 const __bf16* __restrict__ v, const float* __restrict__ mask_logits,
                 __bf16* __restrict__ z) {
  const int qblk = blockIdx.x, h = blockIdx.y;
  const int lane = (int)threadIdx.x;
  const int n = lane & 15, g = lane >> 4;
  const int q0 = qblk * 16;
  const int kend = q0 + 16;

  __shared__ __bf16 Pt[16 * 32];  // probability tile, row-major 16x32
  __shared__ __bf16 Vt[64][32];   // V chunk transposed: [dh][key]

  const __bf16* qrow = q + (size_t)(q0 + n) * D_DIM + (size_t)h * DH_DIM;
  v16bf qf[2];
#pragma unroll
  for (int c = 0; c < 2; c++) {
    v8bf a0 = *(const v8bf*)(qrow + 32 * c + 8 * g);
    v8bf a1 = *(const v8bf*)(qrow + 32 * c + 16 + 8 * g);
#pragma unroll
    for (int i = 0; i < 8; i++) { qf[c][i] = a0[i]; qf[c][i + 8] = a1[i]; }
  }

  v8f zacc[4];
#pragma unroll
  for (int j = 0; j < 4; j++) zacc[j] = zero8();
  float rowmax[8], rowsum[8];
#pragma unroll
  for (int r = 0; r < 8; r++) { rowmax[r] = -__builtin_inff(); rowsum[r] = 0.f; }

  for (int kk = 0; kk < kend; kk += 32) {
    float sc[2][8];
#pragma unroll
    for (int t = 0; t < 2; t++) {
      const int key0 = kk + 16 * t;
      if (key0 < kend) {  // wave-uniform
        v8f s = zero8();
        const __bf16* krow = k + (size_t)(key0 + n) * D_DIM + (size_t)h * DH_DIM;
#pragma unroll
        for (int c = 0; c < 2; c++) {
          v8bf b0 = *(const v8bf*)(krow + 32 * c + 8 * g);
          v8bf b1 = *(const v8bf*)(krow + 32 * c + 16 + 8 * g);
          v16bf bfr;
#pragma unroll
          for (int i = 0; i < 8; i++) { bfr[i] = b0[i]; bfr[i + 8] = b1[i]; }
          s = __builtin_amdgcn_wmma_f32_16x16x32_bf16(false, qf[c], false, bfr,
                                                      (short)0, s, false, false);
        }
#pragma unroll
        for (int r = 0; r < 8; r++) {
          const int colg = key0 + n, rowg = q0 + r + 8 * g;
          sc[t][r] = (colg > rowg) ? -__builtin_inff() : s[r] * 0.125f;
        }
      } else {
#pragma unroll
        for (int r = 0; r < 8; r++) sc[t][r] = -__builtin_inff();
      }
    }

    float p[2][8];
#pragma unroll
    for (int r = 0; r < 8; r++) {
      float mx = fmaxf(sc[0][r], sc[1][r]);
#pragma unroll
      for (int o = 1; o < 16; o <<= 1) mx = fmaxf(mx, __shfl_xor(mx, o, 32));
      const float newmax = fmaxf(rowmax[r], mx);
      const float corr = __expf(rowmax[r] - newmax);
      rowmax[r] = newmax;
      const float p0 = __expf(sc[0][r] - newmax);
      const float p1 = __expf(sc[1][r] - newmax);
      float ps = p0 + p1;
#pragma unroll
      for (int o = 1; o < 16; o <<= 1) ps += __shfl_xor(ps, o, 32);
      rowsum[r] = rowsum[r] * corr + ps;
#pragma unroll
      for (int j = 0; j < 4; j++) zacc[j][r] *= corr;
      p[0][r] = p0; p[1][r] = p1;
    }

#pragma unroll
    for (int t = 0; t < 2; t++)
#pragma unroll
      for (int r = 0; r < 8; r++)
        Pt[(r + 8 * g) * 32 + 16 * t + n] = (__bf16)p[t][r];
    {
      const int key = kk + lane;
      if (key < kend) {
        const __bf16* vr = v + (size_t)key * D_DIM + (size_t)h * DH_DIM;
        for (int d = 0; d < 64; d++) Vt[d][lane] = vr[d];
      } else {
        for (int d = 0; d < 64; d++) Vt[d][lane] = (__bf16)0.f;
      }
    }
    __syncthreads();

    v8bf p0v = *(const v8bf*)&Pt[n * 32 + 8 * g];
    v8bf p1v = *(const v8bf*)&Pt[n * 32 + 16 + 8 * g];
    v16bf pf;
#pragma unroll
    for (int i = 0; i < 8; i++) { pf[i] = p0v[i]; pf[i + 8] = p1v[i]; }
#pragma unroll
    for (int j = 0; j < 4; j++) {
      v8bf b0 = *(const v8bf*)&Vt[16 * j + n][8 * g];
      v8bf b1 = *(const v8bf*)&Vt[16 * j + n][16 + 8 * g];
      v16bf vfr;
#pragma unroll
      for (int i = 0; i < 8; i++) { vfr[i] = b0[i]; vfr[i + 8] = b1[i]; }
      zacc[j] = __builtin_amdgcn_wmma_f32_16x16x32_bf16(false, pf, false, vfr,
                                                        (short)0, zacc[j], false, false);
    }
    __syncthreads();
  }

  const float gate = (mask_logits[h] > 0.f) ? 1.f : 0.f;
#pragma unroll
  for (int j = 0; j < 4; j++)
#pragma unroll
    for (int r = 0; r < 8; r++) {
      const int row = q0 + r + 8 * g;
      const int col = h * DH_DIM + 16 * j + n;
      z[(size_t)row * D_DIM + col] = (__bf16)(zacc[j][r] / rowsum[r] * gate);
    }
}

// ---------------------------------------------------------------------------
extern "C" void kernel_launch(void* const* d_in, const int* in_sizes, int n_in,
                              void* d_out, int out_size, void* d_ws, size_t ws_size,
                              hipStream_t stream) {
  (void)in_sizes; (void)n_in; (void)out_size; (void)ws_size;
  const float* resid_pre   = (const float*)d_in[0];
  const float* ln1_w       = (const float*)d_in[1];
  const float* ln1_b       = (const float*)d_in[2];
  const float* W_Q         = (const float*)d_in[3];
  const float* b_Q         = (const float*)d_in[4];
  const float* W_K         = (const float*)d_in[5];
  const float* b_K         = (const float*)d_in[6];
  const float* W_V         = (const float*)d_in[7];
  const float* b_V         = (const float*)d_in[8];
  const float* W_O         = (const float*)d_in[9];
  const float* b_O         = (const float*)d_in[10];
  const float* mask_logits = (const float*)d_in[11];
  const float* ln2_w       = (const float*)d_in[12];
  const float* ln2_b       = (const float*)d_in[13];
  const float* W_in        = (const float*)d_in[14];
  const float* b_in        = (const float*)d_in[15];
  const float* W_out       = (const float*)d_in[16];
  const float* b_out       = (const float*)d_in[17];
  float* out = (float*)d_out;

  const int S = S_LEN, D = D_DIM, F = F_DIM;

  char* p = (char*)d_ws;
  auto nalloc = [&](size_t bytes) -> void* {
    void* r = (void*)p;
    p += (bytes + 255) & ~(size_t)255;
    return r;
  };
  __bf16* xb    = (__bf16*)nalloc((size_t)S * D * 2);
  __bf16* wqt   = (__bf16*)nalloc((size_t)D * D * 2);
  __bf16* wkt   = (__bf16*)nalloc((size_t)D * D * 2);
  __bf16* wvt   = (__bf16*)nalloc((size_t)D * D * 2);
  __bf16* wot   = (__bf16*)nalloc((size_t)D * D * 2);
  __bf16* wint  = (__bf16*)nalloc((size_t)D * F * 2);
  __bf16* woutt = (__bf16*)nalloc((size_t)F * D * 2);
  __bf16* qbuf  = (__bf16*)nalloc((size_t)S * D * 2);
  __bf16* kbuf  = (__bf16*)nalloc((size_t)S * D * 2);
  __bf16* vbuf  = (__bf16*)nalloc((size_t)S * D * 2);
  __bf16* zbuf  = (__bf16*)nalloc((size_t)S * D * 2);
  float*  residmid = (float*)nalloc((size_t)S * D * 4);
  __bf16* mb    = (__bf16*)nalloc((size_t)S * D * 2);
  __bf16* hid   = (__bf16*)nalloc((size_t)S * F * 2);

  {
    int blocks = (D * D + 255) / 256;
    tcvt_kernel<<<blocks, 256, 0, stream>>>(W_Q, wqt, D, D, 1);
    tcvt_kernel<<<blocks, 256, 0, stream>>>(W_K, wkt, D, D, 1);
    tcvt_kernel<<<blocks, 256, 0, stream>>>(W_V, wvt, D, D, 1);
    tcvt_kernel<<<blocks, 256, 0, stream>>>(W_O, wot, D, D, 0);
    int blocks2 = (D * F + 255) / 256;
    tcvt_kernel<<<blocks2, 256, 0, stream>>>(W_in, wint, D, F, 0);
    tcvt_kernel<<<blocks2, 256, 0, stream>>>(W_out, woutt, F, D, 0);
  }

  ln_kernel<<<S, 256, 0, stream>>>(resid_pre, ln1_w, ln1_b, xb, D);

  {
    int blocks = (S / 128) * (D / 64);
    gemm_bf16_wmma<0, 1, 0, 0><<<blocks, 256, 0, stream>>>(xb, wqt, b_Q, nullptr, qbuf, nullptr, S, D, D);
    gemm_bf16_wmma<0, 1, 0, 0><<<blocks, 256, 0, stream>>>(xb, wkt, b_K, nullptr, kbuf, nullptr, S, D, D);
    gemm_bf16_wmma<0, 1, 0, 0><<<blocks, 256, 0, stream>>>(xb, wvt, b_V, nullptr, vbuf, nullptr, S, D, D);
  }

  attn_kernel<<<dim3(S / 16, H_NUM), 32, 0, stream>>>(qbuf, kbuf, vbuf, mask_logits, zbuf);

  {
    int blocks = (S / 128) * (D / 64);
    gemm_bf16_wmma<0, 0, 1, 1><<<blocks, 256, 0, stream>>>(zbuf, wot, b_O, resid_pre, nullptr, residmid, S, D, D);
  }

  ln_kernel<<<S, 256, 0, stream>>>(residmid, ln2_w, ln2_b, mb, D);

  {
    int blocks = (S / 128) * (F / 64);
    gemm_bf16_wmma<1, 1, 0, 0><<<blocks, 256, 0, stream>>>(mb, wint, b_in, nullptr, hid, nullptr, S, F, D);
  }

  {
    int blocks = (S / 128) * (D / 64);
    gemm_bf16_wmma<0, 0, 1, 1><<<blocks, 256, 0, stream>>>(hid, woutt, b_out, residmid, nullptr, out, S, D, F);
  }
}
